// DroneGAT_42047729828125
// MI455X (gfx1250) — matrix operand order repaired and checked by reading, not compiled, not run on hardware
//
#include <hip/hip_runtime.h>

// ---------------------------------------------------------------------------
// DroneGAT forward for MI455X (gfx1250, wave32, WMMA, async-to-LDS).
//   BN stats (recompute pose MLP) -> last-edge-wins FiLM encode (bf16)
//   -> CSR-by-dst build (histogram + scan + fill)
//   -> GEMM1 bf16 WMMA -> attention projections -> gather aggregation
//      (per-(node,head) wave: max / exp-sum / weighted sum, fused ELU, bf16)
//   -> GEMM2 bf16 WMMA -> projections -> gather aggregation -> d_out (f32).
// Float atomics only in BN stats; GEMM staging uses GLOBAL_LOAD_ASYNC_TO_LDS_B128
// when the toolchain exposes the builtin.
// ---------------------------------------------------------------------------

#define DEVFN __device__ __forceinline__

constexpr int NN    = 65536;
constexpr int EE    = 262144;
constexpr int EP    = EE + NN;   // edges + self loops
constexpr int HID   = 128;
constexpr int HEADS = 8;
constexpr float EPSF = 1e-5f;

typedef __attribute__((ext_vector_type(16))) __bf16 bf16x16;
typedef __attribute__((ext_vector_type(8)))  __bf16 bf16x8;
typedef __attribute__((ext_vector_type(8)))  float  f32x8;

#if defined(__has_builtin)
#if __has_builtin(__builtin_amdgcn_global_load_async_to_lds_b128)
#define HAVE_ASYNC_LDS 1
#endif
#if __has_builtin(__builtin_amdgcn_s_wait_asynccnt)
#define HAVE_WAIT_ASYNC 1
#endif
#endif

DEVFN unsigned short f2bf(float f) {
  unsigned int u = __float_as_uint(f);
  u = (u + 0x7fffu + ((u >> 16) & 1u)) >> 16;   // round-to-nearest-even
  return (unsigned short)u;
}
DEVFN float lrelu02(float x) { return x > 0.f ? x : 0.2f * x; }

// ---------------------------------------------------------------- fills -----
__global__ void fill_i32(int* p, int v, int n) {
  int i = blockIdx.x * blockDim.x + threadIdx.x;
  if (i < n) p[i] = v;
}
__global__ void fill_f32(float* p, float v, int n) {
  int i = blockIdx.x * blockDim.x + threadIdx.x;
  if (i < n) p[i] = v;
}

// --------------------------------------------------------- BN statistics ----
__global__ __launch_bounds__(64)
void bn1_stats(const float* __restrict__ poses, const float* __restrict__ w1,
               const float* __restrict__ b1, float* __restrict__ sum1,
               float* __restrict__ sq1) {
  __shared__ float poseS[64 * 6];
  const int t = threadIdx.x;                      // column 0..63
  float wc[6];
  #pragma unroll
  for (int j = 0; j < 6; ++j) wc[j] = w1[j * 64 + t];
  const float bc = b1[t];
  float s = 0.f, q = 0.f;
  for (int base = blockIdx.x * 64; base < EE; base += gridDim.x * 64) {
    const int ne = min(64, EE - base);
    for (int idx = t; idx < ne * 6; idx += 64) poseS[idx] = poses[base * 6 + idx];
    __syncthreads();
    for (int e = 0; e < ne; ++e) {
      float a = bc;
      #pragma unroll
      for (int j = 0; j < 6; ++j) a += poseS[e * 6 + j] * wc[j];
      s += a; q += a * a;
    }
    __syncthreads();
  }
  atomicAdd(&sum1[t], s);
  atomicAdd(&sq1[t], q);
}

__global__ void bn_finalize(const float* __restrict__ sum, const float* __restrict__ sq,
                            float* __restrict__ mu, float* __restrict__ rstd,
                            int n, float invE) {
  int t = blockIdx.x * blockDim.x + threadIdx.x;
  if (t < n) {
    float m = sum[t] * invE;
    float v = sq[t] * invE - m * m;
    mu[t] = m;
    rstd[t] = rsqrtf(v + EPSF);
  }
}

__global__ __launch_bounds__(128)
void bn2_stats(const float* __restrict__ poses, const float* __restrict__ w1,
               const float* __restrict__ b1, const float* __restrict__ mu1,
               const float* __restrict__ rstd1, const float* __restrict__ g1,
               const float* __restrict__ be1, const float* __restrict__ w2,
               const float* __restrict__ b2, float* __restrict__ sum2,
               float* __restrict__ sq2) {
  __shared__ float poseS[32 * 6];
  __shared__ float h1S[32 * 64];
  const int t = threadIdx.x;                      // column 0..127
  float w2c[64];
  #pragma unroll
  for (int j = 0; j < 64; ++j) w2c[j] = w2[j * HID + t];
  const float b2c = b2[t];
  float s = 0.f, q = 0.f;
  for (int base = blockIdx.x * 32; base < EE; base += gridDim.x * 32) {
    const int ne = min(32, EE - base);
    for (int idx = t; idx < ne * 6; idx += 128) poseS[idx] = poses[base * 6 + idx];
    __syncthreads();
    for (int idx = t; idx < ne * 64; idx += 128) {
      const int e = idx >> 6, j = idx & 63;
      float a = b1[j];
      #pragma unroll
      for (int p = 0; p < 6; ++p) a += poseS[e * 6 + p] * w1[p * 64 + j];
      float h = (a - mu1[j]) * rstd1[j] * g1[j] + be1[j];
      h1S[idx] = fmaxf(h, 0.f);
    }
    __syncthreads();
    for (int e = 0; e < ne; ++e) {
      float a = b2c;
      #pragma unroll
      for (int j = 0; j < 64; ++j) a += h1S[e * 64 + j] * w2c[j];
      s += a; q += a * a;
    }
    __syncthreads();
  }
  atomicAdd(&sum2[t], s);
  atomicAdd(&sq2[t], q);
}

// ------------------------------------------------------- last edge wins -----
__global__ void last_edge(const int* __restrict__ ei, int* __restrict__ last) {
  int e = blockIdx.x * blockDim.x + threadIdx.x;
  if (e < EE) atomicMax(&last[ei[EE + e]], e);
}

// FiLM + scatter: one block (128 thr) per node; emits bf16 (GEMM1 operand).
__global__ __launch_bounds__(128)
void build_enc(const int* __restrict__ last, const float* __restrict__ poses,
               const float* __restrict__ nf, const int* __restrict__ ei,
               const float* __restrict__ w1, const float* __restrict__ b1,
               const float* __restrict__ mu1, const float* __restrict__ rstd1,
               const float* __restrict__ g1, const float* __restrict__ be1,
               const float* __restrict__ w2, const float* __restrict__ b2,
               const float* __restrict__ mu2, const float* __restrict__ rstd2,
               const float* __restrict__ g2, const float* __restrict__ be2,
               const float* __restrict__ gamma_w, const float* __restrict__ gamma_b,
               const float* __restrict__ beta_w, const float* __restrict__ beta_b,
               unsigned short* __restrict__ encb) {
  __shared__ float poseS[6];
  __shared__ float h1S[64];
  __shared__ float h2S[128];
  const int n = blockIdx.x;
  const int t = threadIdx.x;
  const int e = last[n];
  if (e < 0) { encb[(size_t)n * HID + t] = f2bf(nf[(size_t)n * HID + t]); return; }
  if (t < 6) poseS[t] = poses[(size_t)e * 6 + t];
  __syncthreads();
  if (t < 64) {
    float a = b1[t];
    #pragma unroll
    for (int j = 0; j < 6; ++j) a += poseS[j] * w1[j * 64 + t];
    float h = (a - mu1[t]) * rstd1[t] * g1[t] + be1[t];
    h1S[t] = fmaxf(h, 0.f);
  }
  __syncthreads();
  {
    float a = b2[t];
    #pragma unroll
    for (int j = 0; j < 64; ++j) a += h1S[j] * w2[j * HID + t];
    float h = (a - mu2[t]) * rstd2[t] * g2[t] + be2[t];
    h2S[t] = fmaxf(h, 0.f);
  }
  __syncthreads();
  float ga = gamma_b[t], be = beta_b[t];
  #pragma unroll 4
  for (int j = 0; j < HID; ++j) {
    const float hv = h2S[j];
    ga += hv * gamma_w[j * HID + t];
    be += hv * beta_w[j * HID + t];
  }
  ga = 2.f / (1.f + __expf(-ga));
  encb[(size_t)n * HID + t] = f2bf(ga * nf[(size_t)ei[e] * HID + t] + be);
}

// ----------------------------------------------------- CSR by destination ---
__global__ void count_edges(const int* __restrict__ ei, int* __restrict__ counts) {
  int e = blockIdx.x * blockDim.x + threadIdx.x;
  if (e < EE) atomicAdd(&counts[ei[EE + e]], 1);
}

__global__ __launch_bounds__(256)
void scan_block(const int* __restrict__ counts, int* __restrict__ excl,
                int* __restrict__ bsum) {
  __shared__ int s[256];
  const int t = threadIdx.x;
  const int g = blockIdx.x * 256 + t;
  const int v = counts[g];
  s[t] = v;
  __syncthreads();
  for (int off = 1; off < 256; off <<= 1) {
    int x = (t >= off) ? s[t - off] : 0;
    __syncthreads();
    s[t] += x;
    __syncthreads();
  }
  excl[g] = s[t] - v;
  if (t == 255) bsum[blockIdx.x] = s[255];
}

__global__ __launch_bounds__(256)
void scan_bsum(int* __restrict__ bsum) {   // 1 block, exclusive in place
  __shared__ int s[256];
  const int t = threadIdx.x;
  const int v = bsum[t];
  s[t] = v;
  __syncthreads();
  for (int off = 1; off < 256; off <<= 1) {
    int x = (t >= off) ? s[t - off] : 0;
    __syncthreads();
    s[t] += x;
    __syncthreads();
  }
  bsum[t] = s[t] - v;
}

__global__ void scan_add(int* __restrict__ excl, const int* __restrict__ bsum) {
  const int g = blockIdx.x * 256 + threadIdx.x;
  excl[g] += bsum[blockIdx.x];
}

__global__ void fill_elist(const int* __restrict__ ei, const int* __restrict__ rowptr,
                           int* __restrict__ cur, int* __restrict__ elist) {
  int e = blockIdx.x * blockDim.x + threadIdx.x;
  if (e >= EP) return;
  const int d = (e < EE) ? ei[EE + e] : (e - EE);
  const int slot = rowptr[d] + atomicAdd(&cur[d], 1);
  elist[slot] = e;
}

// ----------------------------------------- weight transpose + bf16 convert --
// BT[n*K + k] = bf16(B[k*Ntot + n])
__global__ void convert_wT(const float* __restrict__ B, unsigned short* __restrict__ BT,
                           int K, int Ntot) {
  const long long idx = (long long)blockIdx.x * blockDim.x + threadIdx.x;
  if (idx >= (long long)K * Ntot) return;
  const int n = (int)(idx / K), k = (int)(idx % K);
  BT[idx] = f2bf(B[(size_t)k * Ntot + n]);
}

// --------------------------------------------- bf16 WMMA tiled GEMM ---------
// C[M,Ntot] = A[M,K] @ B[K,Ntot]; A row-major bf16, BT = B^T row-major bf16.
// Block tile 128x128, k-chunk 64, 8 waves: each wave 64x32 (4x2 WMMA tiles).
#define BM 128
#define BN 128
#define BK 64
#define BKP 72    // padded LDS row stride (halfwords), keeps 16B alignment

// Builtin expects int-vector pointers (per hipcc diagnostic):
//   src: addrspace(1) int4-vector*, dst: addrspace(3) int4-vector*
typedef __attribute__((__vector_size__(4 * sizeof(int)))) int i32x4v;
#define ASG(p) ((__attribute__((address_space(1))) i32x4v*)(p))
#define ASL(p) ((__attribute__((address_space(3))) i32x4v*)(p))

__global__ __launch_bounds__(256)
void gemm_bf16_wmma(const unsigned short* __restrict__ A,
                    const unsigned short* __restrict__ BT,
                    float* __restrict__ C, int M, int K, int Ntot) {
  __shared__ __align__(32) unsigned short lsA[BM * BKP];
  __shared__ __align__(32) unsigned short lsB[BN * BKP];
  const int tid  = threadIdx.x;
  const int lane = tid & 31;
  const int wv   = tid >> 5;
  const int wm   = wv >> 2;                  // 0..1 -> 64-row half
  const int wn   = wv & 3;                   // 0..3 -> 32-col quarter
  const size_t rowBase = (size_t)blockIdx.y * BM;
  const size_t colBase = (size_t)blockIdx.x * BN;

  const f32x8 zero = {0.f, 0.f, 0.f, 0.f, 0.f, 0.f, 0.f, 0.f};
  f32x8 acc[4][2];
  #pragma unroll
  for (int mt = 0; mt < 4; ++mt)
    #pragma unroll
    for (int nt = 0; nt < 2; ++nt) acc[mt][nt] = zero;

  const int steps = K / BK;
  for (int s = 0; s < steps; ++s) {
    const int k0 = s * BK;
    __syncthreads();
#ifdef HAVE_ASYNC_LDS
    // Async DMA: 16B per lane per op, tracked by ASYNCcnt (no VALU staging).
    for (int c = tid; c < BM * 8; c += 256) {
      const int m = c >> 3, kc = (c & 7) * 8;
      __builtin_amdgcn_global_load_async_to_lds_b128(
          ASG((void*)&A[(rowBase + m) * (size_t)K + k0 + kc]),
          ASL(&lsA[m * BKP + kc]), 0, 0);
    }
    for (int c = tid; c < BN * 8; c += 256) {
      const int n = c >> 3, kc = (c & 7) * 8;
      __builtin_amdgcn_global_load_async_to_lds_b128(
          ASG((void*)&BT[(colBase + n) * (size_t)K + k0 + kc]),
          ASL(&lsB[n * BKP + kc]), 0, 0);
    }
#ifdef HAVE_WAIT_ASYNC
    __builtin_amdgcn_s_wait_asynccnt(0);
#else
    asm volatile("s_wait_asynccnt 0x0" ::: "memory");
#endif
#else
    for (int c = tid; c < BM * 8; c += 256) {
      const int m = c >> 3, kc = (c & 7) * 8;
      *(uint4*)&lsA[m * BKP + kc] =
          *(const uint4*)&A[(rowBase + m) * (size_t)K + k0 + kc];
    }
    for (int c = tid; c < BN * 8; c += 256) {
      const int n = c >> 3, kc = (c & 7) * 8;
      *(uint4*)&lsB[n * BKP + kc] =
          *(const uint4*)&BT[(colBase + n) * (size_t)K + k0 + kc];
    }
#endif
    __syncthreads();

    const int h  = lane >> 4;     // half-wave
    const int ml = lane & 15;     // M row / N col within 16
    #pragma unroll
    for (int kk = 0; kk < BK; kk += 32) {
      bf16x16 bfrag[2];
      #pragma unroll
      for (int nt = 0; nt < 2; ++nt) {
        const unsigned short* p = &lsB[(wn * 32 + nt * 16 + ml) * BKP + kk + 16 * h];
        bf16x8 b0 = *(const bf16x8*)p;
        bf16x8 b1 = *(const bf16x8*)(p + 8);
        bfrag[nt] = __builtin_shufflevector(b0, b1, 0, 1, 2, 3, 4, 5, 6, 7,
                                            8, 9, 10, 11, 12, 13, 14, 15);
      }
      #pragma unroll
      for (int mt = 0; mt < 4; ++mt) {
        const unsigned short* p = &lsA[(wm * 64 + mt * 16 + ml) * BKP + kk + 16 * h];
        bf16x8 a0 = *(const bf16x8*)p;
        bf16x8 a1 = *(const bf16x8*)(p + 8);
        bf16x16 afrag = __builtin_shufflevector(a0, a1, 0, 1, 2, 3, 4, 5, 6, 7,
                                                8, 9, 10, 11, 12, 13, 14, 15);
        #pragma unroll
        for (int nt = 0; nt < 2; ++nt)
          acc[mt][nt] = __builtin_amdgcn_wmma_f32_16x16x32_bf16(
              false, afrag, false, bfrag[nt], (short)0, acc[mt][nt], false, false);
      }
    }
  }

  // Epilogue: C/D layout — lane(nl=lane&15, h=lane>>4), vgpr r -> row r+8h.
  const int h  = lane >> 4;
  const int nl = lane & 15;
  #pragma unroll
  for (int mt = 0; mt < 4; ++mt)
    #pragma unroll
    for (int nt = 0; nt < 2; ++nt) {
      const size_t row0 = rowBase + wm * 64 + mt * 16 + 8 * h;
      const size_t col  = colBase + wn * 32 + nt * 16 + nl;
      #pragma unroll
      for (int r = 0; r < 8; ++r)
        C[(row0 + r) * (size_t)Ntot + col] = acc[mt][nt][r];
    }
}

// -------------------------------------------- attention logit projections ---
__global__ __launch_bounds__(256)
void attn_prep(const float* __restrict__ xw, const float* __restrict__ a_src,
               const float* __restrict__ a_dst, float* __restrict__ als,
               float* __restrict__ ald, int heads) {
  const int lane = threadIdx.x & 31;
  const int node = blockIdx.x * 8 + (threadIdx.x >> 5);
  if (node >= NN) return;
  const int W = heads * HID;
  for (int hd = 0; hd < heads; ++hd) {
    float s = 0.f, d = 0.f;
    #pragma unroll
    for (int c = lane; c < HID; c += 32) {
      const float x = xw[(size_t)node * W + hd * HID + c];
      s += x * a_src[hd * HID + c];
      d += x * a_dst[hd * HID + c];
    }
    #pragma unroll
    for (int off = 16; off > 0; off >>= 1) {
      s += __shfl_xor(s, off, 32);
      d += __shfl_xor(d, off, 32);
    }
    if (lane == 0) { als[node * heads + hd] = s; ald[node * heads + hd] = d; }
  }
}

// ------------------------- gather-style GAT aggregation (no float atomics) --
// One wave per (node, head): pass 1 = max logit, pass 2 = exp-sum + weighted
// gather of xw[src] with channels across lanes; fused bias (+ELU, bf16 out).
__global__ __launch_bounds__(256)
void gat_aggregate(const int* __restrict__ ei, const int* __restrict__ rowptr,
                   const int* __restrict__ elist, const float* __restrict__ als,
                   const float* __restrict__ ald, const float* __restrict__ xw,
                   const float* __restrict__ bias, float* __restrict__ outF,
                   unsigned short* __restrict__ outB, int heads, int elu) {
  const int lane = threadIdx.x & 31;
  const long long widx = (long long)blockIdx.x * 8 + (threadIdx.x >> 5);
  if (widx >= (long long)NN * heads) return;
  const int n  = (int)(widx / heads);
  const int hd = (int)(widx % heads);
  const int beg = rowptr[n];
  const int end = (n == NN - 1) ? EP : rowptr[n + 1];   // >= beg+1 (self loop)
  const float aldv = ald[n * heads + hd];
  const int W = heads * HID;

  // Pass 1: stable-softmax max, lane-parallel over the edge list.
  float mx = -3.402823466e38f;
  for (int i = beg + lane; i < end; i += 32) {
    const int e = elist[i];
    const int s = (e < EE) ? ei[e] : (e - EE);
    mx = fmaxf(mx, lrelu02(als[s * heads + hd] + aldv));
  }
  #pragma unroll
  for (int off = 16; off > 0; off >>= 1) mx = fmaxf(mx, __shfl_xor(mx, off, 32));

  // Pass 2: serial over edges, 128 channels across lanes (4 per lane).
  float den = 0.f, a0 = 0.f, a1 = 0.f, a2 = 0.f, a3 = 0.f;
  for (int i = beg; i < end; ++i) {
    const int e = elist[i];
    const int s = (e < EE) ? ei[e] : (e - EE);
    const float ex = __expf(lrelu02(als[s * heads + hd] + aldv) - mx);
    den += ex;
    const float* xs = xw + (size_t)s * W + hd * HID;
    a0 += ex * xs[lane];
    a1 += ex * xs[32 + lane];
    a2 += ex * xs[64 + lane];
    a3 += ex * xs[96 + lane];
  }
  const float inv = 1.f / den;      // den identical across lanes
  const float v[4] = {a0, a1, a2, a3};
  #pragma unroll
  for (int c4 = 0; c4 < 4; ++c4) {
    const int col = hd * HID + c4 * 32 + lane;
    float x = v[c4] * inv + bias[col];
    if (elu) x = x > 0.f ? x : (__expf(x) - 1.f);
    if (outB) outB[(size_t)n * W + col] = f2bf(x);
    else      outF[(size_t)n * W + col] = x;
  }
}

// ---------------------------------------------------------------------------
extern "C" void kernel_launch(void* const* d_in, const int* in_sizes, int n_in,
                              void* d_out, int out_size, void* d_ws, size_t ws_size,
                              hipStream_t stream) {
  const float* nf      = (const float*)d_in[0];
  const float* poses   = (const float*)d_in[1];
  const float* pe_w1   = (const float*)d_in[2];
  const float* pe_b1   = (const float*)d_in[3];
  const float* pe_g1   = (const float*)d_in[4];
  const float* pe_be1  = (const float*)d_in[5];
  const float* pe_w2   = (const float*)d_in[6];
  const float* pe_b2   = (const float*)d_in[7];
  const float* pe_g2   = (const float*)d_in[8];
  const float* pe_be2  = (const float*)d_in[9];
  const float* gamma_w = (const float*)d_in[10];
  const float* gamma_b = (const float*)d_in[11];
  const float* beta_w  = (const float*)d_in[12];
  const float* beta_b  = (const float*)d_in[13];
  const float* g1_w    = (const float*)d_in[14];
  const float* g1_as   = (const float*)d_in[15];
  const float* g1_ad   = (const float*)d_in[16];
  const float* g1_b    = (const float*)d_in[17];
  const float* g2_w    = (const float*)d_in[18];
  const float* g2_as   = (const float*)d_in[19];
  const float* g2_ad   = (const float*)d_in[20];
  const float* g2_b    = (const float*)d_in[21];
  const int*   ei      = (const int*)d_in[22];
  float* out = (float*)d_out;
  char*  ws  = (char*)d_ws;

  // ---- workspace layout ----
  float* sum1  = (float*)ws;                   // 64
  float* sq1   = sum1 + 64;                    // 64
  float* sum2  = sq1 + 64;                     // 128
  float* sq2   = sum2 + 128;                   // 128
  float* mu1   = sq2 + 128;                    // 64
  float* rstd1 = mu1 + 64;                     // 64
  float* mu2   = rstd1 + 64;                   // 128
  float* rstd2 = mu2 + 128;                    // 128
  size_t off = 4096;
  int*   last   = (int*)(ws + off);   off += (size_t)NN * 4;
  int*   counts = (int*)(ws + off);   off += (size_t)NN * 4;   // reused as cursor
  int*   rowptr = (int*)(ws + off);   off += (size_t)NN * 4;
  int*   elist  = (int*)(ws + off);   off += (size_t)EP * 4;
  int*   bsum   = (int*)(ws + off);   off += 4096;
  float* als1   = (float*)(ws + off); off += (size_t)NN * HEADS * 4;
  float* ald1   = (float*)(ws + off); off += (size_t)NN * HEADS * 4;
  float* als2   = (float*)(ws + off); off += (size_t)NN * 4;
  float* ald2   = (float*)(ws + off); off += (size_t)NN * 4;
  unsigned short* encb = (unsigned short*)(ws + off); off += (size_t)NN * HID * 2;
  unsigned short* g1wT = (unsigned short*)(ws + off); off += (size_t)HID * 1024 * 2;
  unsigned short* g2wT = (unsigned short*)(ws + off); off += (size_t)1024 * HID * 2;
  float* xw1  = (float*)(ws + off);           off += (size_t)NN * 1024 * 4;
  unsigned short* x2b = (unsigned short*)(ws + off); off += (size_t)NN * 1024 * 2;
  float* xw2  = (float*)(ws + off);           off += (size_t)NN * HID * 4;
  if (ws_size < off) return;

  auto blocks = [](long long n, int bs) { return (unsigned)((n + bs - 1) / bs); };

  // ---- init ----
  fill_f32<<<blocks(704, 256), 256, 0, stream>>>((float*)ws, 0.f, 704);
  fill_i32<<<blocks(NN, 256), 256, 0, stream>>>(last, -1, NN);
  fill_i32<<<blocks(NN, 256), 256, 0, stream>>>(counts, 1, NN);   // self loop

  // ---- BatchNorm statistics (recompute pose MLP, no [E,*] storage) ----
  bn1_stats<<<1024, 64, 0, stream>>>(poses, pe_w1, pe_b1, sum1, sq1);
  bn_finalize<<<1, 64, 0, stream>>>(sum1, sq1, mu1, rstd1, 64, 1.f / EE);
  bn2_stats<<<1024, 128, 0, stream>>>(poses, pe_w1, pe_b1, mu1, rstd1, pe_g1, pe_be1,
                                      pe_w2, pe_b2, sum2, sq2);
  bn_finalize<<<1, 128, 0, stream>>>(sum2, sq2, mu2, rstd2, 128, 1.f / EE);

  // ---- last-edge-wins scatter + FiLM encode (bf16) ----
  last_edge<<<blocks(EE, 256), 256, 0, stream>>>(ei, last);
  build_enc<<<NN, 128, 0, stream>>>(last, poses, nf, ei, pe_w1, pe_b1, mu1, rstd1,
                                    pe_g1, pe_be1, pe_w2, pe_b2, mu2, rstd2, pe_g2,
                                    pe_be2, gamma_w, gamma_b, beta_w, beta_b, encb);

  // ---- CSR by destination (histogram -> 2-level scan -> fill) ----
  count_edges<<<blocks(EE, 256), 256, 0, stream>>>(ei, counts);
  scan_block<<<NN / 256, 256, 0, stream>>>(counts, rowptr, bsum);
  scan_bsum<<<1, 256, 0, stream>>>(bsum);
  scan_add<<<NN / 256, 256, 0, stream>>>(rowptr, bsum);
  fill_i32<<<blocks(NN, 256), 256, 0, stream>>>(counts, 0, NN);   // cursor
  fill_elist<<<blocks(EP, 256), 256, 0, stream>>>(ei, rowptr, counts, elist);

  // ---- weight transpose + bf16 convert ----
  convert_wT<<<blocks((long long)HID * 1024, 256), 256, 0, stream>>>(g1_w, g1wT,
                                                                     HID, 1024);
  convert_wT<<<blocks((long long)1024 * HID, 256), 256, 0, stream>>>(g2_w, g2wT,
                                                                     1024, HID);

  // ---- GAT layer 1 ----
  gemm_bf16_wmma<<<dim3(1024 / BN, NN / BM), 256, 0, stream>>>(encb, g1wT, xw1,
                                                               NN, HID, 1024);
  attn_prep<<<NN / 8, 256, 0, stream>>>(xw1, g1_as, g1_ad, als1, ald1, HEADS);
  gat_aggregate<<<blocks((long long)NN * HEADS, 8), 256, 0, stream>>>(
      ei, rowptr, elist, als1, ald1, xw1, g1_b, nullptr, x2b, HEADS, 1);

  // ---- GAT layer 2 ----
  gemm_bf16_wmma<<<dim3(HID / BN, NN / BM), 256, 0, stream>>>(x2b, g2wT, xw2,
                                                              NN, 1024, HID);
  attn_prep<<<NN / 8, 256, 0, stream>>>(xw2, g2_as, g2_ad, als2, ald2, 1);
  gat_aggregate<<<blocks(NN, 8), 256, 0, stream>>>(
      ei, rowptr, elist, als2, ald2, xw2, g2_b, out, nullptr, 1, 0);
}
